// CAMD_32693291057810
// MI455X (gfx1250) — matrix operand order repaired and checked by reading, not compiled, not run on hardware
//
#include <hip/hip_runtime.h>
#include <hip/hip_bf16.h>
#include <cstddef>
#include <cstdint>

typedef __bf16 bf16;
typedef __attribute__((ext_vector_type(16))) __bf16 v16bf;
typedef __attribute__((ext_vector_type(8)))  __bf16 v8bf;
typedef __attribute__((ext_vector_type(8)))  float  v8f;
typedef __attribute__((ext_vector_type(4)))  unsigned v4u;
typedef __attribute__((ext_vector_type(8)))  int      v8i;
typedef __attribute__((ext_vector_type(4)))  int      v4i;

#if __has_builtin(__builtin_amdgcn_tensor_load_to_lds) && \
    __has_builtin(__builtin_amdgcn_s_wait_tensorcnt)
#define HAVE_TDM 1
#else
#define HAVE_TDM 0
#endif

__device__ __forceinline__ v8f wmma_bf16(v16bf a, v16bf b, v8f c) {
  // D = A(16x32 bf16) * B(32x16 bf16) + C(16x16 f32)
  return __builtin_amdgcn_wmma_f32_16x16x32_bf16(false, a, false, b, (short)0, c,
                                                 false, false);
}

__device__ __forceinline__ v16bf concat8(v8bf lo, v8bf hi8) {
  return __builtin_shufflevector(lo, hi8, 0, 1, 2, 3, 4, 5, 6, 7,
                                 8, 9, 10, 11, 12, 13, 14, 15);
}

// A fragment (16xK chunk, K=32) from row-major bf16 LDS [rows x ldm].
// element e -> K = kbase + 16*(e>>3) + 8*hi + (e&7): two 16B runs -> 2x ds_load_b128
__device__ __forceinline__ v16bf frag_A(const bf16* src, int ldm, int row, int hi,
                                        int kbase) {
  const bf16* p = src + row * ldm + kbase + 8 * hi;
  v8bf lo = *(const v8bf*)p;
  v8bf hb = *(const v8bf*)(p + 16);
  return concat8(lo, hb);
}

// B fragment (32x16) where B[k][n] = srcT[(nbase+n)*ldm + kbase + k].
// element e -> k = 16*(e>>3) + 8*hi + (e&7): two 16B runs -> 2x ds_load_b128
__device__ __forceinline__ v16bf frag_BT(const bf16* src, int ldm, int n, int hi,
                                         int kbase, int nbase) {
  const bf16* p = src + (nbase + n) * ldm + kbase + 8 * hi;
  v8bf lo = *(const v8bf*)p;
  v8bf hb = *(const v8bf*)(p + 16);
  return concat8(lo, hb);
}

__device__ __forceinline__ unsigned pack2bf(float a, float b) {
  union { bf16 h[2]; unsigned u; } u;
  u.h[0] = (bf16)a;
  u.h[1] = (bf16)b;
  return u.u;
}

#if HAVE_TDM
// 1-D TDM copy: nbytes (multiple of 8, <= 512KB) from global to LDS.
// D# per ISA 8.3/8.4: group0 = {count, lds_addr, global_addr, type=2},
// group1 = {data_size=8B, tensor_dim0 = tile_dim0 = nbytes/8, tensor_dim1=1}.
__device__ __forceinline__ void tdm_load_1d(const void* gsrc, void* ldst,
                                            unsigned nbytes) {
  unsigned lds_addr = (unsigned)(uintptr_t)ldst;  // low 32b of flat = LDS offset
  unsigned long long ga = (unsigned long long)(uintptr_t)gsrc;
  unsigned n8 = nbytes >> 3;  // 8-byte elements
  v4u g0;
  g0[0] = 1u;                                   // count=1 (valid user D#)
  g0[1] = lds_addr;                             // lds_addr
  g0[2] = (unsigned)ga;                         // global_addr[31:0]
  g0[3] = (unsigned)(ga >> 32) | (2u << 30);    // global_addr[56:32] | type=2
  v8i g1;
  g1[0] = (int)(3u << 16);                      // data_size=3 (8B); wg_mask=0
  g1[1] = (int)((n8 & 0xFFFFu) << 16);          // tensor_dim0[15:0]  (bits 63:48)
  g1[2] = (int)(((n8 >> 16) & 0xFFFFu) | (1u << 16));  // td0[31:16] | tensor_dim1=1
  g1[3] = (int)((n8 & 0xFFFFu) << 16);          // tile_dim0 (bits 127:112)
  g1[4] = 0;                                    // tile_dim1=0, tile_dim2=0
  g1[5] = (int)n8;                              // tensor_dim0_stride[31:0]
  g1[6] = 0;
  g1[7] = 0;
  v4i gz = {0, 0, 0, 0};
#if defined(__clang_major__) && __clang_major__ >= 23
  v8i gz8 = {0, 0, 0, 0, 0, 0, 0, 0};
  __builtin_amdgcn_tensor_load_to_lds(g0, g1, gz, gz, gz8, 0);
#else
  __builtin_amdgcn_tensor_load_to_lds(g0, g1, gz, gz, 0);
#endif
}
#endif

// ---------------------------------------------------------------------------
// 3-layer MLP (64->64->64->64, relu, relu, linear), bf16 WMMA, f32 accum.
// Block = 128 threads (4 waves); each block owns 64 rows, each wave 16 rows.
// ---------------------------------------------------------------------------
#define WT_LDM 72  // 144B rows: 16B aligned, spreads LDS banks

__global__ void mlp64_wmma(const float* __restrict__ X, const float* __restrict__ W,
                           const float* __restrict__ B, bf16* __restrict__ Out) {
  __shared__ __align__(16) bf16 Abuf[64 * 64];
  __shared__ __align__(16) bf16 WbufT[64 * WT_LDM];
  __shared__ float bias[64];

  const int tid = threadIdx.x;
  const int wave = tid >> 5, lane = tid & 31;
  const int hi = lane >> 4, n = lane & 15;
  const int rowbase = blockIdx.x * 64;
  const int qb = wave * 16;

  // stage input activations: float4 loads -> packed bf16 b64 stores
  {
    const float4* X4 = (const float4*)(X + (size_t)rowbase * 64);
    uint2* A64 = (uint2*)Abuf;
    for (int i = tid; i < 64 * 64 / 4; i += 128) {
      float4 x = X4[i];
      A64[i] = make_uint2(pack2bf(x.x, x.y), pack2bf(x.z, x.w));
    }
  }

  for (int layer = 0; layer < 3; ++layer) {
    __syncthreads();
    const float* Wl = W + (size_t)layer * 64 * 64;
    // transpose W into LDS: WbufT[n][k] = W[k][n] (b128 global reads)
    {
      const float4* W4 = (const float4*)Wl;
      for (int i = tid; i < 64 * 16; i += 128) {
        int k = i >> 4, n4 = (i & 15) * 4;
        float4 w = W4[i];
        WbufT[(n4 + 0) * WT_LDM + k] = (bf16)w.x;
        WbufT[(n4 + 1) * WT_LDM + k] = (bf16)w.y;
        WbufT[(n4 + 2) * WT_LDM + k] = (bf16)w.z;
        WbufT[(n4 + 3) * WT_LDM + k] = (bf16)w.w;
      }
    }
    if (tid < 64) bias[tid] = B[layer * 64 + tid];
    __syncthreads();

    v16bf a0 = frag_A(Abuf, 64, qb + n, hi, 0);
    v16bf a1 = frag_A(Abuf, 64, qb + n, hi, 32);

    v8f acc[4];
#pragma unroll
    for (int nt = 0; nt < 4; ++nt) {
      v16bf b0 = frag_BT(WbufT, WT_LDM, n, hi, 0, nt * 16);
      v16bf b1 = frag_BT(WbufT, WT_LDM, n, hi, 32, nt * 16);
      v8f c = {};
      c = wmma_bf16(a0, b0, c);
      c = wmma_bf16(a1, b1, c);
      acc[nt] = c;
    }
    __syncthreads();

#pragma unroll
    for (int nt = 0; nt < 4; ++nt) {
#pragma unroll
      for (int v = 0; v < 8; ++v) {
        int r = qb + v + 8 * hi;   // row within 64-row tile (wave-private)
        int cc = nt * 16 + n;      // output feature
        float val = acc[nt][v] + bias[cc];
        if (layer < 2) val = fmaxf(val, 0.0f);
        if (layer == 2)
          Out[(size_t)(rowbase + r) * 64 + cc] = (bf16)val;
        else
          Abuf[r * 64 + cc] = (bf16)val;
      }
    }
  }
}

// ---------------------------------------------------------------------------
// Timestamp-causal linear attention:
//   partial[mod][i, 0:32] = sum_{j: t2[j] <= t1[i]} (Q_i . K_j) * X[j, 0:32]
// Block = 128 threads (4 waves x 16 queries). grid = (T1/64, 4 modalities).
// K chunks are streamed into double-buffered LDS by the Tensor Data Mover
// (wave 0 issues, TENSORcnt synchronizes), overlapping DMA with WMMA work.
// T2 is always a multiple of 32 and jb+32 <= T2, so chunk loads are unguarded.
// ---------------------------------------------------------------------------
#define VT_LDM 40  // 80B rows, 16B aligned

__global__ void attn_wmma(const bf16* __restrict__ Qbf, const float* __restrict__ M1,
                          const bf16* K0, const bf16* K1, const bf16* K2, const bf16* K3,
                          const float* X0, const float* X1, const float* X2, const float* X3,
                          int t20, int t21, int t22, int t23,
                          float* __restrict__ partial) {
  __shared__ __align__(16) bf16 Qs[64 * 64];
  __shared__ __align__(16) bf16 Kt[2][32 * 64];  // double-buffered K chunk
  __shared__ __align__(16) bf16 VtT[32 * VT_LDM];  // VtT[col][key]
  __shared__ float t1s[64];
  __shared__ float t2s[32];
  __shared__ __align__(16) bf16 Sbuf[4][16 * 32];  // masked scores, per wave
  __shared__ int nj_s;

  const int tid = threadIdx.x;
  const int wave = tid >> 5, lane = tid & 31;
  const int hi = lane >> 4, n = lane & 15;
  const int qstart = blockIdx.x * 64;
  const int qb = wave * 16;

  const int mod = blockIdx.y;
  const bf16* Kb;
  const float* X;
  int T2;
  switch (mod) {
    case 0: Kb = K0; X = X0; T2 = t20; break;
    case 1: Kb = K1; X = X1; T2 = t21; break;
    case 2: Kb = K2; X = X2; T2 = t22; break;
    default: Kb = K3; X = X3; T2 = t23; break;
  }
  float* Out = partial + (size_t)mod * (size_t)gridDim.x * 64 * 32;

  if (tid < 64) t1s[tid] = M1[(size_t)(qstart + tid) * 64 + 63];
  {  // stage Q tile as b128 copies
    const uint4* qsrc = (const uint4*)(Qbf + (size_t)qstart * 64);
    uint4* qdst = (uint4*)Qs;
    for (int i = tid; i < 64 * 64 * 2 / 16; i += 128) qdst[i] = qsrc[i];
  }
  __syncthreads();

  if (tid == 0) {  // keys with t2 <= max t1 of this block (both sorted)
    float t1max = t1s[63];
    int lo = 0, hib = T2;
    while (lo < hib) {
      int mid = (lo + hib) >> 1;
      float t = X[(size_t)mid * 64 + 63];
      if (t <= t1max) lo = mid + 1; else hib = mid;
    }
    nj_s = lo;
  }
  __syncthreads();
  const int nj = nj_s;  // block-uniform
  const int nchunks = (nj + 31) >> 5;

  v16bf aq0 = frag_A(Qs, 64, qb + n, hi, 0);
  v16bf aq1 = frag_A(Qs, 64, qb + n, hi, 32);
  float tqv[8];
#pragma unroll
  for (int v = 0; v < 8; ++v) tqv[v] = t1s[qb + v + 8 * hi];

  v8f o0 = {}, o1 = {};
  bf16* Sw = &Sbuf[wave][0];

#if HAVE_TDM
  if (nchunks > 0 && wave == 0)  // prologue: DMA chunk 0 into Kt[0]
    tdm_load_1d(Kb, &Kt[0][0], 32 * 64 * 2);
#endif

  for (int c = 0; c < nchunks; ++c) {
    const int jb = c * 32;
    const int cur = c & 1;
    __syncthreads();  // retire previous iteration's readers of VtT/t2s/Kt[cur^1]

#if !HAVE_TDM
    {  // fallback: cooperative b128 copy of K chunk
      const uint4* ksrc = (const uint4*)(Kb + (size_t)jb * 64);
      uint4* kdst = (uint4*)&Kt[cur][0];
      for (int i = tid; i < 32 * 64 * 2 / 16; i += 128) kdst[i] = ksrc[i];
    }
#endif
    // V chunk transposed: VtT[col][key] = X[jb+key][col], cols 0..31 (b128 reads)
    {
      const float4* X4 = (const float4*)X;
      for (int i = tid; i < 32 * 8; i += 128) {
        int j = i >> 3, c4 = (i & 7) * 4;
        float4 x = X4[(size_t)(jb + j) * 16 + (i & 7)];
        VtT[(c4 + 0) * VT_LDM + j] = (bf16)x.x;
        VtT[(c4 + 1) * VT_LDM + j] = (bf16)x.y;
        VtT[(c4 + 2) * VT_LDM + j] = (bf16)x.z;
        VtT[(c4 + 3) * VT_LDM + j] = (bf16)x.w;
      }
    }
    if (tid < 32) t2s[tid] = X[(size_t)(jb + tid) * 64 + 63];

#if HAVE_TDM
    if (wave == 0) __builtin_amdgcn_s_wait_tensorcnt(0);  // Kt[cur] complete
#endif
    __syncthreads();  // Kt[cur] + VtT + t2s visible to all waves
#if HAVE_TDM
    if (wave == 0 && c + 1 < nchunks)  // overlap next chunk's DMA with compute
      tdm_load_1d(Kb + (size_t)(jb + 32) * 64, &Kt[cur ^ 1][0], 32 * 64 * 2);
#endif
    const bf16* Kc = &Kt[cur][0];

    // S = Q * K^T for two 16-key subtiles; mask by timestamps; spill bf16
#pragma unroll
    for (int sj = 0; sj < 2; ++sj) {
      v16bf b0 = frag_BT(Kc, 64, n, hi, 0, sj * 16);
      v16bf b1 = frag_BT(Kc, 64, n, hi, 32, sj * 16);
      v8f s = {};
      s = wmma_bf16(aq0, b0, s);
      s = wmma_bf16(aq1, b1, s);
      float tk = t2s[sj * 16 + n];
#pragma unroll
      for (int v = 0; v < 8; ++v)
        Sw[(v + 8 * hi) * 32 + sj * 16 + n] = (bf16)((tk <= tqv[v]) ? s[v] : 0.0f);
    }
    __syncthreads();

    // O += S(16x32) * V(32x32): S fragment is A-pattern -> two b128 loads
    v16bf as = frag_A(Sw, 32, n, hi, 0);
    v16bf bv0 = frag_BT(VtT, VT_LDM, n, hi, 0, 0);
    v16bf bv1 = frag_BT(VtT, VT_LDM, n, hi, 0, 16);
    o0 = wmma_bf16(as, bv0, o0);
    o1 = wmma_bf16(as, bv1, o1);
  }

#pragma unroll
  for (int v = 0; v < 8; ++v) {
    int r = qstart + qb + v + 8 * hi;
    Out[(size_t)r * 32 + n] = o0[v];
    Out[(size_t)r * 32 + 16 + n] = o1[v];
  }
}

__global__ void reduce4_kernel(const float* __restrict__ p, float* __restrict__ out,
                               int n) {
  int i = blockIdx.x * blockDim.x + threadIdx.x;
  if (i < n) out[i] = p[i] + p[n + i] + p[2 * n + i] + p[3 * n + i];
}

extern "C" void kernel_launch(void* const* d_in, const int* in_sizes, int n_in,
                              void* d_out, int out_size, void* d_ws, size_t ws_size,
                              hipStream_t stream) {
  (void)n_in; (void)out_size; (void)ws_size;
  const float* X[4] = {(const float*)d_in[0], (const float*)d_in[1],
                       (const float*)d_in[2], (const float*)d_in[3]};
  const float* WQw = (const float*)d_in[4];
  const float* WQb = (const float*)d_in[5];
  const float* WKw = (const float*)d_in[6];
  const float* WKb = (const float*)d_in[7];

  int T[4];
  for (int i = 0; i < 4; ++i) T[i] = in_sizes[i] / 64;
  const int T1 = T[0];

  // workspace carve (256B aligned): Q bf16, 4x K bf16, 4x partial f32
  char* ws = (char*)d_ws;
  size_t off = 0;
  auto carve = [&](size_t bytes) {
    char* p = ws + off;
    off += (bytes + 255) & ~(size_t)255;
    return p;
  };
  bf16* Qbf = (bf16*)carve((size_t)T1 * 64 * sizeof(bf16));
  bf16* Kbf[4];
  for (int m = 0; m < 4; ++m) Kbf[m] = (bf16*)carve((size_t)T[m] * 64 * sizeof(bf16));
  float* partial = (float*)carve((size_t)4 * T1 * 32 * sizeof(float));

  // Q = mlp(m1, WQ); K_m = mlp(X_m, WK[m])
  mlp64_wmma<<<dim3(T1 / 64), dim3(128), 0, stream>>>(X[0], WQw, WQb, Qbf);
  for (int m = 0; m < 4; ++m)
    mlp64_wmma<<<dim3(T[m] / 64), dim3(128), 0, stream>>>(
        X[m], WKw + (size_t)m * 3 * 64 * 64, WKb + (size_t)m * 3 * 64, Kbf[m]);

  attn_wmma<<<dim3(T1 / 64, 4), dim3(128), 0, stream>>>(
      Qbf, X[0], Kbf[0], Kbf[1], Kbf[2], Kbf[3],
      X[0], X[1], X[2], X[3], T[0], T[1], T[2], T[3], partial);

  const int ntot = T1 * 32;
  reduce4_kernel<<<dim3((ntot + 255) / 256), dim3(256), 0, stream>>>(
      partial, (float*)d_out, ntot);
}
